// GraphAttentionEmbedding_3075196584340
// MI455X (gfx1250) — matrix operand order repaired and checked by reading, not compiled, not run on hardware
//
#include <hip/hip_runtime.h>
#include <hip/hip_bf16.h>

typedef __attribute__((ext_vector_type(16))) _Float16 v16h;
typedef __attribute__((ext_vector_type(8)))  _Float16 v8h;
typedef __attribute__((ext_vector_type(8)))  float    v8f;

#define DIM 128      // IN == OUT == 128
#define MSGD 64
#define TD 32
#define NEG_FLT_MAX (-3.4028234663852886e38f)

// ---------------------------------------------------------------------------
// float atomic max via int trick (works for mixed signs; init = -FLT_MAX)
__device__ inline void atomicMaxF(float* addr, float v) {
    if (v >= 0.0f) atomicMax((int*)addr, __float_as_int(v));
    else           atomicMin((unsigned int*)addr, (unsigned int)__float_as_int(v));
}

// ---------------------------------------------------------------------------
// One 16x16 output tile of A[16x128] @ B[128x16], A in LDS row-major f16,
// B column-major f16 (Bt[n*128 + k]).  CDNA5 wave32 WMMA layouts:
//   A frag: lane L: row = L%16, half = L/16; elems 0..7 = K 8*half+j,
//           elems 8..15 = K 16+8*half+j  (per k-tile of 32)
//   B frag: lane L: col = L%16, same K order as A frag
__device__ inline v8f wmma_row128(const _Float16* As, const _Float16* Bt,
                                  int lane, v8f acc) {
    const int rm = lane & 15;
    const int half = lane >> 4;
    const int n = lane & 15;
#pragma unroll
    for (int kt = 0; kt < 4; ++kt) {
        const int kb = kt * 32;
        union { v16h v; v8h h[2]; } a;
        a.h[0] = *(const v8h*)(As + rm * DIM + kb + 8 * half);
        a.h[1] = *(const v8h*)(As + rm * DIM + kb + 16 + 8 * half);
        union { v16h v; v8h h[2]; } bu;
        bu.h[0] = *(const v8h*)(Bt + n * DIM + kb + 8 * half);
        bu.h[1] = *(const v8h*)(Bt + n * DIM + kb + 16 + 8 * half);
        v16h b = bu.v;
        acc = __builtin_amdgcn_wmma_f32_16x16x32_f16(
            false, a.v, false, b, (short)0, acc, false, false);
    }
    return acc;
}

// ---------------------------------------------------------------------------
__global__ __launch_bounds__(256)
void pack_x_kernel(const float* __restrict__ x, _Float16* __restrict__ xh, int n) {
    for (int i = blockIdx.x * 256 + threadIdx.x; i < n; i += gridDim.x * 256)
        xh[i] = (_Float16)x[i];
}

// Transpose-convert Wq,Wk,Wv,Wskip ([k][n] -> [n][k]) and We (pad K 96->128)
__global__ __launch_bounds__(256)
void pack_w_kernel(const float* __restrict__ Wq, const float* __restrict__ Wk,
                   const float* __restrict__ Wv, const float* __restrict__ Ws,
                   const float* __restrict__ We,
                   _Float16* __restrict__ Wt, _Float16* __restrict__ Wet) {
    int i = blockIdx.x * 256 + threadIdx.x;
    if (i < 4 * DIM * DIM) {
        int mat = i >> 14, rem = i & (DIM * DIM - 1);
        int n = rem >> 7, k = rem & 127;
        const float* W = (mat == 0) ? Wq : (mat == 1) ? Wk : (mat == 2) ? Wv : Ws;
        Wt[i] = (_Float16)W[k * DIM + n];
    } else if (i < 5 * DIM * DIM) {
        int rem = i - 4 * DIM * DIM;
        int n = rem >> 7, k = rem & 127;
        Wet[rem] = (_Float16)((k < 96) ? We[k * DIM + n] : 0.0f);
    }
}

__global__ __launch_bounds__(256)
void init_kernel(float* __restrict__ mg, float* __restrict__ dn, int n2) {
    int i = blockIdx.x * 256 + threadIdx.x;
    if (i < n2) { mg[i] = NEG_FLT_MAX; dn[i] = 0.0f; }
}

// ---------------------------------------------------------------------------
// Node projections: q/k/v (f16 tables) + skip (f32, written to d_out)
__global__ __launch_bounds__(256)
void proj_kernel(const _Float16* __restrict__ xh, const _Float16* __restrict__ Wt,
                 const float* __restrict__ bq, const float* __restrict__ bk,
                 const float* __restrict__ bv, const float* __restrict__ bs,
                 _Float16* __restrict__ qh, _Float16* __restrict__ kh,
                 _Float16* __restrict__ vh, float* __restrict__ out, int N) {
    __shared__ __align__(32) _Float16 As[16 * DIM];
    const int tile = blockIdx.x, t = threadIdx.x;
    {   // stage 16 rows of x (f16), zero-pad past N
        int row = t >> 4, chunk = t & 15;
        int g = tile * 16 + row;
        v8h val = {};
        if (g < N) val = *(const v8h*)(xh + (size_t)g * DIM + chunk * 8);
        *(v8h*)(As + row * DIM + chunk * 8) = val;
    }
    __syncthreads();
    const int wave = t >> 5, lane = t & 31;
    const int colbase = wave * 16, half = lane >> 4, n = lane & 15;
    v8f aq = {}, ak = {}, av = {}, as = {};
    aq = wmma_row128(As, Wt + 0 * DIM * DIM + colbase * DIM, lane, aq);
    ak = wmma_row128(As, Wt + 1 * DIM * DIM + colbase * DIM, lane, ak);
    av = wmma_row128(As, Wt + 2 * DIM * DIM + colbase * DIM, lane, av);
    as = wmma_row128(As, Wt + 3 * DIM * DIM + colbase * DIM, lane, as);
    const int col = colbase + n;
    const float Bq = bq[col], Bk = bk[col], Bv = bv[col], Bs = bs[col];
#pragma unroll
    for (int r = 0; r < 8; ++r) {            // D layout: M = r + 8*half, N = col
        int g = tile * 16 + r + 8 * half;
        if (g < N) {
            size_t o = (size_t)g * DIM + col;
            qh[o] = (_Float16)(aq[r] + Bq);
            kh[o] = (_Float16)(ak[r] + Bk);
            vh[o] = (_Float16)(av[r] + Bv);
            out[o] = as[r] + Bs;
        }
    }
}

// ---------------------------------------------------------------------------
// Build a 16-edge x 128 padded edge_attr tile in LDS (f16)
__device__ inline void build_attr(_Float16* attr, int* srcs, int* dsts, float* relt,
                                  const int* __restrict__ ei,
                                  const float* __restrict__ lu,
                                  const float* __restrict__ tarr,
                                  const float* __restrict__ msg,
                                  const float* __restrict__ tw,
                                  const float* __restrict__ tb,
                                  int tile, int E, int t) {
    if (t < 16) {
        int eg = tile * 16 + t;
        int s = 0, d = 0; float rt = 0.0f;
        if (eg < E) { s = ei[eg]; d = ei[E + eg]; rt = lu[s] - tarr[eg]; }
        srcs[t] = s; dsts[t] = d; relt[t] = rt;
    }
    __syncthreads();
    for (int i = t; i < 16 * TD; i += 256) {            // time encoding
        int m = i >> 5, c = i & 31;
        attr[m * DIM + c] = (_Float16)__cosf(relt[m] * tw[c] + tb[c]);
    }
    for (int i = t; i < 16 * MSGD; i += 256) {          // msg
        int m = i >> 6, c = i & 63;
        int eg = tile * 16 + m;
        attr[m * DIM + TD + c] = (_Float16)((eg < E) ? msg[(size_t)eg * MSGD + c] : 0.0f);
    }
    for (int i = t; i < 16 * 32; i += 256) {            // zero pad K 96..127
        int m = i >> 5, c = i & 31;
        attr[m * DIM + 96 + c] = (_Float16)0.0f;
    }
    __syncthreads();
}

// ---------------------------------------------------------------------------
// Pass A: e = attr@We, alpha = dot(q[dst], k[src]+e)/8, segment max
__global__ __launch_bounds__(256)
void edge_alpha_kernel(const int* __restrict__ ei, const float* __restrict__ lu,
                       const float* __restrict__ tarr, const float* __restrict__ msg,
                       const float* __restrict__ tw, const float* __restrict__ tb,
                       const _Float16* __restrict__ Wet,
                       const _Float16* __restrict__ qh, const _Float16* __restrict__ kh,
                       float* __restrict__ alpha_out, float* __restrict__ mg, int E) {
    __shared__ __align__(32) _Float16 attr[16 * DIM];
    __shared__ float e_f[16 * DIM];
    __shared__ float alpha_s[32];
    __shared__ int srcs[16], dsts[16];
    __shared__ float relt[16];
    const int tile = blockIdx.x, t = threadIdx.x;
    if (t < 32) alpha_s[t] = 0.0f;
    build_attr(attr, srcs, dsts, relt, ei, lu, tarr, msg, tw, tb, tile, E, t);

    const int wave = t >> 5, lane = t & 31;
    const int colbase = wave * 16, half = lane >> 4, n = lane & 15;
    v8f acc = {};
    acc = wmma_row128(attr, Wet + colbase * DIM, lane, acc);
#pragma unroll
    for (int r = 0; r < 8; ++r)
        e_f[(r + 8 * half) * DIM + colbase + n] = acc[r];
    __syncthreads();

    {   // 256 threads: (m, h, part-of-8) partial dot products
        int m = t & 15, h = (t >> 4) & 1, part = t >> 5;
        int s = srcs[m], d = dsts[m];
        const _Float16* qp = qh + (size_t)d * DIM + h * 64 + part * 8;
        const _Float16* kp = kh + (size_t)s * DIM + h * 64 + part * 8;
        const float*    ep = e_f + m * DIM + h * 64 + part * 8;
        float p = 0.0f;
#pragma unroll
        for (int j = 0; j < 8; ++j) p += (float)qp[j] * ((float)kp[j] + ep[j]);
        atomicAdd(&alpha_s[h * 16 + m], p);
    }
    __syncthreads();
    if (t < 32) {
        int m = t & 15, h = t >> 4;
        int eg = tile * 16 + m;
        if (eg < E) {
            float a = alpha_s[h * 16 + m] * 0.125f;   // 1/sqrt(64)
            alpha_out[(size_t)eg * 2 + h] = a;
            atomicMaxF(&mg[(size_t)dsts[m] * 2 + h], a);
        }
    }
}

// ---------------------------------------------------------------------------
// Pass B: ex = exp(alpha - m[dst]); denom += ex
__global__ __launch_bounds__(256)
void denom_kernel(const float* __restrict__ alpha, const int* __restrict__ ei,
                  const float* __restrict__ mg, float* __restrict__ ex_out,
                  float* __restrict__ dn, int E) {
    int i = blockIdx.x * 256 + threadIdx.x;
    if (i < E * 2) {
        int eg = i >> 1, h = i & 1;
        int d = ei[E + eg];
        float ex = __expf(alpha[i] - mg[(size_t)d * 2 + h]);
        ex_out[i] = ex;
        atomicAdd(&dn[(size_t)d * 2 + h], ex);
    }
}

// ---------------------------------------------------------------------------
// Pass C: recompute e, out[dst] += (v[src] + e) * a
__global__ __launch_bounds__(256)
void edge_out_kernel(const int* __restrict__ ei, const float* __restrict__ lu,
                     const float* __restrict__ tarr, const float* __restrict__ msg,
                     const float* __restrict__ tw, const float* __restrict__ tb,
                     const _Float16* __restrict__ Wet, const _Float16* __restrict__ vh,
                     const float* __restrict__ ex, const float* __restrict__ dn,
                     float* __restrict__ out, int E) {
    __shared__ __align__(32) _Float16 attr[16 * DIM];
    __shared__ float a_s[32];
    __shared__ int srcs[16], dsts[16];
    __shared__ float relt[16];
    const int tile = blockIdx.x, t = threadIdx.x;
    build_attr(attr, srcs, dsts, relt, ei, lu, tarr, msg, tw, tb, tile, E, t);
    if (t < 32) {
        int m = t & 15, h = t >> 4;
        int eg = tile * 16 + m;
        float a = 0.0f;
        if (eg < E) a = ex[(size_t)eg * 2 + h] / (dn[(size_t)dsts[m] * 2 + h] + 1e-16f);
        a_s[h * 16 + m] = a;
    }
    __syncthreads();

    const int wave = t >> 5, lane = t & 31;
    const int colbase = wave * 16, half = lane >> 4, n = lane & 15;
    v8f acc = {};
    acc = wmma_row128(attr, Wet + colbase * DIM, lane, acc);
    const int col = colbase + n, h = col >> 6;
#pragma unroll
    for (int r = 0; r < 8; ++r) {
        int m = r + 8 * half;
        int eg = tile * 16 + m;
        if (eg < E) {
            float val = ((float)vh[(size_t)srcs[m] * DIM + col] + acc[r]) * a_s[h * 16 + m];
            atomicAdd(&out[(size_t)dsts[m] * DIM + col], val);
        }
    }
}

// ---------------------------------------------------------------------------
extern "C" void kernel_launch(void* const* d_in, const int* in_sizes, int n_in,
                              void* d_out, int out_size, void* d_ws, size_t ws_size,
                              hipStream_t stream) {
    const float* x    = (const float*)d_in[0];
    const float* lu   = (const float*)d_in[1];
    const float* tarr = (const float*)d_in[2];
    const float* msg  = (const float*)d_in[3];
    const int*   ei   = (const int*)d_in[4];
    const float* tw   = (const float*)d_in[5];
    const float* tb   = (const float*)d_in[6];
    const float* Wq   = (const float*)d_in[7];
    const float* bq   = (const float*)d_in[8];
    const float* Wk   = (const float*)d_in[9];
    const float* bk   = (const float*)d_in[10];
    const float* Wv   = (const float*)d_in[11];
    const float* bv   = (const float*)d_in[12];
    const float* We   = (const float*)d_in[13];
    const float* Ws   = (const float*)d_in[14];
    const float* bs   = (const float*)d_in[15];
    float* out = (float*)d_out;

    const int N = in_sizes[1];   // last_update
    const int E = in_sizes[2];   // t

    char* ws = (char*)d_ws;
    size_t o = 0;
    auto alloc = [&](size_t bytes) { size_t r = o; o += (bytes + 255) & ~(size_t)255; return r; };
    _Float16* xh  = (_Float16*)(ws + alloc((size_t)N * DIM * 2));
    _Float16* qh  = (_Float16*)(ws + alloc((size_t)N * DIM * 2));
    _Float16* kh  = (_Float16*)(ws + alloc((size_t)N * DIM * 2));
    _Float16* vh  = (_Float16*)(ws + alloc((size_t)N * DIM * 2));
    _Float16* Wt  = (_Float16*)(ws + alloc((size_t)4 * DIM * DIM * 2));
    _Float16* Wet = (_Float16*)(ws + alloc((size_t)DIM * DIM * 2));
    float* alpha  = (float*)(ws + alloc((size_t)E * 2 * 4));
    float* exv    = (float*)(ws + alloc((size_t)E * 2 * 4));
    float* mg     = (float*)(ws + alloc((size_t)N * 2 * 4));
    float* dn     = (float*)(ws + alloc((size_t)N * 2 * 4));

    pack_x_kernel<<<4096, 256, 0, stream>>>(x, xh, N * DIM);
    pack_w_kernel<<<(5 * DIM * DIM + 255) / 256, 256, 0, stream>>>(Wq, Wk, Wv, Ws, We, Wt, Wet);
    init_kernel<<<(2 * N + 255) / 256, 256, 0, stream>>>(mg, dn, 2 * N);
    proj_kernel<<<(N + 15) / 16, 256, 0, stream>>>(xh, Wt, bq, bk, bv, bs, qh, kh, vh, out, N);
    edge_alpha_kernel<<<(E + 15) / 16, 256, 0, stream>>>(ei, lu, tarr, msg, tw, tb, Wet, qh, kh,
                                                         alpha, mg, E);
    denom_kernel<<<(2 * E + 255) / 256, 256, 0, stream>>>(alpha, ei, mg, exv, dn, E);
    edge_out_kernel<<<(E + 15) / 16, 256, 0, stream>>>(ei, lu, tarr, msg, tw, tb, Wet, vh,
                                                       exv, dn, out, E);
}
